// LSTM_55327768708454
// MI455X (gfx1250) — matrix operand order repaired and checked by reading, not compiled, not run on hardware
//
#include <hip/hip_runtime.h>
#include <hip/hip_bf16.h>

// ---------------------------------------------------------------------------
// LSTM forward for MI455X (gfx1250, wave32, WMMA).
// GEMMs: bf16 WMMA (f32 accum), double-buffered LDS tiles filled with
// async global->LDS DMA (ASYNCcnt) when the toolchain exposes the builtin.
// ---------------------------------------------------------------------------

typedef __bf16 bf16;
typedef __attribute__((ext_vector_type(8)))  __bf16 v8bf;
typedef __attribute__((ext_vector_type(16))) __bf16 v16bf;
typedef __attribute__((ext_vector_type(8)))  float  v8f;

constexpr int kT  = 128;
constexpr int kB  = 256;
constexpr int kD  = 512;
constexpr int kH  = 1024;
constexpr int kFH = 4096;   // 4*H
constexpr int kTB = kT * kB;

// Padded LDS row stride (halfs): 32 data + 8 pad -> per-lane ds_load_b128
// start banks are distinct multiples of 4 dwords => conflict-free.
constexpr int kPad = 40;

#if defined(__AMDGCN__) && __has_builtin(__builtin_amdgcn_global_load_async_to_lds_b128)
#define HAS_ASYNC_LDS 1
#else
#define HAS_ASYNC_LDS 0
#endif

union U16bf { v16bf v; v8bf h[2]; };

// A-fragment (16x32 bf16, MxK) per ISA 7.12.2. p already includes +hi*8;
// halfs 0..7 at p[0..7], halfs 8..15 at p[16..23].
__device__ __forceinline__ v16bf load_frag_a(const bf16* p) {
  U16bf u;
  u.h[0] = *(const v8bf*)(p);
  u.h[1] = *(const v8bf*)(p + 16);
  return u.v;
}

__device__ __forceinline__ v8f wmma_bf16(v16bf a, v16bf b, v8f c) {
  return __builtin_amdgcn_wmma_f32_16x16x32_bf16(
      false, a, false, b, (short)0, c, false, false);
}

// One 16-byte global->LDS transfer. Async DMA path uses ASYNCcnt; the
// fallback is a plain global load + ds_store (still staged through LDS).
__device__ __forceinline__ void cp16_g2l(bf16* lds, const bf16* g) {
#if HAS_ASYNC_LDS
  // Builtin signature (from diagnostics): (v4i addrspace(1)*, v4i addrspace(3)*, imm, imm)
  typedef __attribute__((__vector_size__(16))) int v4i;
  typedef __attribute__((address_space(1))) v4i* gp_t;
  typedef __attribute__((address_space(3))) v4i* lp_t;
  __builtin_amdgcn_global_load_async_to_lds_b128(
      (gp_t)(g), (lp_t)(lds), 0, 0);
#else
  *(v8bf*)lds = *(const v8bf*)g;
#endif
}

__device__ __forceinline__ void wait_async_3() {
#if HAS_ASYNC_LDS
#if __has_builtin(__builtin_amdgcn_s_wait_asynccnt)
  __builtin_amdgcn_s_wait_asynccnt(3);
#else
  asm volatile("s_wait_asynccnt 0x3" ::: "memory");
#endif
#endif
}

__device__ __forceinline__ void wait_async_0() {
#if HAS_ASYNC_LDS
#if __has_builtin(__builtin_amdgcn_s_wait_asynccnt)
  __builtin_amdgcn_s_wait_asynccnt(0);
#else
  asm volatile("s_wait_asynccnt 0x0" ::: "memory");
#endif
#endif
}

// C[M,N] = A[M,K] * B[N,K]^T (+ Cin)(+ bias1 + bias2)(, relu)
// Block: 256 threads = 8 waves (2 M x 4 N), block tile 64 x 128,
// wave tile 32 x 32 = 2x2 WMMA accumulators, K stepped by 32.
// A/B tiles staged in double-buffered LDS; each thread issues exactly
// 3 b128 copies per tile (1 for A's 64x32, 2 for B's 128x32).
__global__ __launch_bounds__(256) void gemm_bf16_wmma(
    const bf16* __restrict__ A, const bf16* __restrict__ B,
    const float* __restrict__ Cin,
    const float* __restrict__ bias1, const float* __restrict__ bias2,
    float* __restrict__ Cf, bf16* __restrict__ Cb,
    int M, int N, int K, int relu)
{
  __shared__ bf16 lA[2][64  * kPad];   // 10 KB
  __shared__ bf16 lB[2][128 * kPad];   // 20 KB

  const int tid  = threadIdx.x;
  const int lane = tid & 31;
  const int wave = tid >> 5;           // 0..7
  const int wm   = wave >> 2;          // 0..1
  const int wn   = wave & 3;           // 0..3
  const int l    = lane & 15;
  const int hi   = lane >> 4;

  const int mblk = blockIdx.y * 64;
  const int nblk = blockIdx.x * 128;

  // Cooperative tile copy mapping: row = tid>>2, 16B segment = tid&3.
  const int crow = tid >> 2;           // 0..63
  const int cseg = (tid & 3) * 8;      // half offset within 64B row chunk

  const bf16* gA  = A + (size_t)(mblk + crow)       * K + cseg;
  const bf16* gB0 = B + (size_t)(nblk + crow)       * K + cseg;
  const bf16* gB1 = B + (size_t)(nblk + 64 + crow)  * K + cseg;
  bf16* sA  = &lA[0][crow        * kPad + cseg];
  bf16* sB0 = &lB[0][crow        * kPad + cseg];
  bf16* sB1 = &lB[0][(64 + crow) * kPad + cseg];
  const size_t bufA = 64  * kPad;      // halfs between the two A buffers
  const size_t bufB = 128 * kPad;

  auto issue_tile = [&](int k0, int buf) {
    __builtin_prefetch(gA  + k0 + 128, 0, 0);   // global_prefetch_b8
    __builtin_prefetch(gB0 + k0 + 128, 0, 0);
    cp16_g2l(sA  + (size_t)buf * bufA, gA  + k0);
    cp16_g2l(sB0 + (size_t)buf * bufB, gB0 + k0);
    cp16_g2l(sB1 + (size_t)buf * bufB, gB1 + k0);
  };

  v8f acc00 = {}, acc01 = {}, acc10 = {}, acc11 = {};
  const int nk = K >> 5;

  issue_tile(0, 0);

  for (int kk = 0; kk < nk; ++kk) {
    const int cur = kk & 1;
    if (kk + 1 < nk) {
      issue_tile((kk + 1) << 5, cur ^ 1);
      wait_async_3();                 // current tile's 3 ops done; next 3 fly
    } else {
      wait_async_0();
    }
    __syncthreads();                  // current buffer visible to all waves

    const bf16* pa0 = &lA[cur][(wm * 32 + l) * kPad + hi * 8];
    const bf16* pa1 = pa0 + 16 * kPad;
    const bf16* pb0 = &lB[cur][(wn * 32 + l) * kPad + hi * 8];
    const bf16* pb1 = pb0 + 16 * kPad;

    v16bf a0 = load_frag_a(pa0);
    v16bf a1 = load_frag_a(pa1);
    v16bf b0 = load_frag_a(pb0);
    v16bf b1 = load_frag_a(pb1);

    acc00 = wmma_bf16(a0, b0, acc00);
    acc01 = wmma_bf16(a0, b1, acc01);
    acc10 = wmma_bf16(a1, b0, acc10);
    acc11 = wmma_bf16(a1, b1, acc11);

    __syncthreads();                  // all reads done before buffer reuse
  }

  // C/D layout: lane l, VGPR r -> (row = 8*hi + r, col = l) in a 16x16 tile.
  const int m0 = mblk + wm * 32;
  const int n0 = nblk + wn * 32;
  v8f accs[2][2] = { { acc00, acc01 }, { acc10, acc11 } };
#pragma unroll
  for (int i = 0; i < 2; ++i) {
#pragma unroll
    for (int j = 0; j < 2; ++j) {
      const int col = n0 + j * 16 + l;
      float badd = 0.f;
      if (bias1) badd += bias1[col];
      if (bias2) badd += bias2[col];
#pragma unroll
      for (int r = 0; r < 8; ++r) {
        const int row = m0 + i * 16 + hi * 8 + r;
        const size_t idx = (size_t)row * N + col;
        float v = accs[i][j][r] + badd;
        if (Cin)  v += Cin[idx];
        if (relu) v = fmaxf(v, 0.f);
        if (Cf) Cf[idx] = v;
        if (Cb) Cb[idx] = (bf16)v;
      }
    }
  }
}

__global__ void cast_f32_to_bf16(const float* __restrict__ in,
                                 bf16* __restrict__ out, int n) {
  int i = blockIdx.x * blockDim.x + threadIdx.x;
  if (i < n) out[i] = (bf16)in[i];
}

__global__ void init_state(const float* __restrict__ h0,
                           const float* __restrict__ c0,
                           float* __restrict__ hf, float* __restrict__ cf,
                           bf16* __restrict__ hb, int n) {
  int i = blockIdx.x * blockDim.x + threadIdx.x;
  if (i < n) {
    float h = h0[i], c = c0[i];
    hf[i] = h; cf[i] = c; hb[i] = (bf16)h;
  }
}

// Gates in PyTorch order i,f,g,o; update c,h; emit bf16 h for next-step GEMM.
__global__ void lstm_gates(const float* __restrict__ g,
                           float* __restrict__ cf, float* __restrict__ hf,
                           bf16* __restrict__ hb, bf16* __restrict__ hs_t,
                           int H, int fourH) {
  int idx = blockIdx.x * blockDim.x + threadIdx.x;  // b*H + j
  int b = idx / H, j = idx - b * H;
  const float* gr = g + (size_t)b * fourH;
  float gi = gr[j];
  float gf = gr[j + H];
  float gg = gr[j + 2 * H];
  float go = gr[j + 3 * H];
  gi = 1.f / (1.f + __expf(-gi));
  gf = 1.f / (1.f + __expf(-gf));
  go = 1.f / (1.f + __expf(-go));
  gg = tanhf(gg);
  float c = gf * cf[idx] + gi * gg;
  float h = go * tanhf(c);
  cf[idx] = c;
  hf[idx] = h;
  hb[idx] = (bf16)h;
  hs_t[idx] = (bf16)h;
}

// out[row] = dot(hs[row,:], W_out) + b_out ; one wave32 per row.
__global__ void out_proj(const bf16* __restrict__ hs,
                         const float* __restrict__ Wout,
                         const float* __restrict__ bout,
                         float* __restrict__ out, int H) {
  int row  = blockIdx.x * (blockDim.x >> 5) + (threadIdx.x >> 5);
  int lane = threadIdx.x & 31;
  const bf16* hr = hs + (size_t)row * H;
  float s = 0.f;
  for (int j = lane; j < H; j += 32) s += (float)hr[j] * Wout[j];
#pragma unroll
  for (int m = 16; m; m >>= 1) s += __shfl_xor(s, m, 32);
  if (lane == 0) out[row] = s + bout[0];
}

__global__ void copy_states(const float* __restrict__ hf,
                            const float* __restrict__ cf,
                            float* __restrict__ outH,
                            float* __restrict__ outC, int n) {
  int i = blockIdx.x * blockDim.x + threadIdx.x;
  if (i < n) { outH[i] = hf[i]; outC[i] = cf[i]; }
}

extern "C" void kernel_launch(void* const* d_in, const int* in_sizes, int n_in,
                              void* d_out, int out_size, void* d_ws, size_t ws_size,
                              hipStream_t stream) {
  (void)in_sizes; (void)n_in; (void)out_size; (void)ws_size;

  const float* inputs = (const float*)d_in[0];
  const float* h0     = (const float*)d_in[1];
  const float* c0     = (const float*)d_in[2];
  const float* W1     = (const float*)d_in[3];
  const float* b1     = (const float*)d_in[4];
  const float* W_ih   = (const float*)d_in[5];
  const float* W_hh   = (const float*)d_in[6];
  const float* b_ih   = (const float*)d_in[7];
  const float* b_hh   = (const float*)d_in[8];
  const float* W_out  = (const float*)d_in[9];
  const float* b_out  = (const float*)d_in[10];

  char* ws = (char*)d_ws;
  size_t off = 0;
  auto alloc = [&](size_t bytes) -> void* {
    void* p = ws + off;
    off = (off + bytes + 255) & ~(size_t)255;
    return p;
  };

  bf16*  Ab   = (bf16*) alloc((size_t)kTB * kD  * 2);  // inputs bf16
  bf16*  W1b  = (bf16*) alloc((size_t)kH  * kD  * 2);
  bf16*  Wihb = (bf16*) alloc((size_t)kFH * kH  * 2);
  bf16*  Whhb = (bf16*) alloc((size_t)kFH * kH  * 2);
  bf16*  Xb   = (bf16*) alloc((size_t)kTB * kH  * 2);  // relu(x) bf16
  float* XG   = (float*)alloc((size_t)kTB * kFH * 4);  // xg f32
  bf16*  Hb   = (bf16*) alloc((size_t)kB  * kH  * 2);  // h bf16 (GEMM input)
  float* Hf   = (float*)alloc((size_t)kB  * kH  * 4);  // h f32
  float* Cst  = (float*)alloc((size_t)kB  * kH  * 4);  // c f32
  float* Gb   = (float*)alloc((size_t)kB  * kFH * 4);  // per-step gates f32
  bf16*  HS   = (bf16*) alloc((size_t)kTB * kH  * 2);  // h history bf16

  // ---- precision casts -----------------------------------------------------
  cast_f32_to_bf16<<<(kTB * kD + 255) / 256, 256, 0, stream>>>(inputs, Ab, kTB * kD);
  cast_f32_to_bf16<<<(kH  * kD + 255) / 256, 256, 0, stream>>>(W1,   W1b,  kH * kD);
  cast_f32_to_bf16<<<(kFH * kH + 255) / 256, 256, 0, stream>>>(W_ih, Wihb, kFH * kH);
  cast_f32_to_bf16<<<(kFH * kH + 255) / 256, 256, 0, stream>>>(W_hh, Whhb, kFH * kH);
  init_state<<<(kB * kH + 255) / 256, 256, 0, stream>>>(h0, c0, Hf, Cst, Hb, kB * kH);

  // ---- GEMM1: x = relu(inputs * W1^T + b1)  (32768x1024, K=512) -----------
  gemm_bf16_wmma<<<dim3(kH / 128, kTB / 64), 256, 0, stream>>>(
      Ab, W1b, nullptr, b1, nullptr, nullptr, Xb, kTB, kH, kD, 1);

  // ---- GEMM2: xg = x * W_ih^T + b_ih + b_hh (32768x4096, K=1024) ----------
  gemm_bf16_wmma<<<dim3(kFH / 128, kTB / 64), 256, 0, stream>>>(
      Xb, Wihb, nullptr, b_ih, b_hh, XG, nullptr, kTB, kFH, kH, 0);

  // ---- recurrence: per step g = xg[t] + h*W_hh^T, then gates --------------
  for (int t = 0; t < kT; ++t) {
    gemm_bf16_wmma<<<dim3(kFH / 128, kB / 64), 256, 0, stream>>>(
        Hb, Whhb, XG + (size_t)t * kB * kFH, nullptr, nullptr,
        Gb, nullptr, kB, kFH, kH, 0);
    lstm_gates<<<(kB * kH) / 256, 256, 0, stream>>>(
        Gb, Cst, Hf, Hb, HS + (size_t)t * kB * kH, kH, kFH);
  }

  // ---- outputs -------------------------------------------------------------
  float* out = (float*)d_out;
  out_proj<<<kTB / 8, 256, 0, stream>>>(HS, W_out, b_out, out, kH);
  copy_states<<<(kB * kH) / 256, 256, 0, stream>>>(
      Hf, Cst, out + kTB, out + kTB + kB * kH, kB * kH);
}
